// WanAttentionBlock_6511170421055
// MI455X (gfx1250) — matrix operand order repaired and checked by reading, not compiled, not run on hardware
//
#include <hip/hip_runtime.h>
#include <hip/hip_bf16.h>

typedef __attribute__((ext_vector_type(16))) _Float16 v16h;
typedef __attribute__((ext_vector_type(8)))  _Float16 v8h;
typedef __attribute__((ext_vector_type(4)))  _Float16 v4h;
typedef __attribute__((ext_vector_type(8)))  float    v8f;
typedef __attribute__((ext_vector_type(4)))  unsigned int u32x4;
typedef __attribute__((ext_vector_type(8)))  unsigned int u32x8;
typedef _Float16 f16;

#define NHEADS 12
#define HD 128
#define LN_EPS 1e-6f

// ---------------------------------------------------------------------------
// WMMA fragment loaders (16x16x32 f16; A and B both use contiguous k = 16*(lane/16) + s)
// ---------------------------------------------------------------------------
__device__ __forceinline__ v16h a_frag_from(const f16* p, int stride) {
    int l = threadIdx.x & 31;
    const f16* r = p + (size_t)(l & 15) * stride + ((l >> 4) * 16);
    v8h lo = *(const v8h*)(r);
    v8h hi = *(const v8h*)(r + 8);
    v16h o;
#pragma unroll
    for (int i = 0; i < 8; i++) { o[i] = lo[i]; o[i + 8] = hi[i]; }
    return o;
}

// storage is [N][K] row-major (n-major), contract over K: contiguous per lane
__device__ __forceinline__ v16h b_frag_nmajor(const f16* p, int stride) {
    int l = threadIdx.x & 31;
    return *(const v16h*)(p + (size_t)(l & 15) * stride + (l >> 4) * 16);
}

// ---------------------------------------------------------------------------
// Tensor Data Mover: 2-D tile load Global->LDS (ISA ch.8 D# descriptor).
// rowlen elements/row (data_size units), nrows = tile rows, avail = rows in
// tensor from tile start (OOB rows read as zero), stride = elements between
// rows. Optional LDS row padding: pad_amt DWORD-code after every
// (1<<pad_int) 8-byte units.
// ---------------------------------------------------------------------------
__device__ __forceinline__ void tdm_load_2d(unsigned lds_off, const void* gaddr,
                                            unsigned rowlen, unsigned nrows,
                                            unsigned avail, unsigned stride_elts,
                                            unsigned dsz_code, unsigned pad_en,
                                            unsigned pad_int, unsigned pad_amt) {
    unsigned long long ga = (unsigned long long)(uintptr_t)gaddr;
    u32x4 g0;
    g0[0] = 1u;                                                // count=1, user mode
    g0[1] = lds_off;                                           // lds_addr (bytes)
    g0[2] = (unsigned)(ga & 0xffffffffu);                      // global_addr[31:0]
    g0[3] = (unsigned)((ga >> 32) & 0x01ffffffu) | (2u << 30); // addr[56:32] | type=2
    u32x8 g1;
    g1[0] = ((dsz_code & 3u) << 16) | ((pad_en & 1u) << 20) |
            ((pad_int & 7u) << 22) | ((pad_amt & 0x7fu) << 25);
    g1[1] = (rowlen & 0xffffu) << 16;                          // tensor_dim0 lo16 @63:48
    g1[2] = ((rowlen >> 16) & 0xffffu) | ((avail & 0xffffu) << 16); // d0 hi | tensor_dim1 lo
    g1[3] = ((avail >> 16) & 0xffffu) | ((rowlen & 0xffffu) << 16); // d1 hi | tile_dim0
    g1[4] = (nrows & 0xffffu);                                 // tile_dim1 (tile_dim2=0)
    g1[5] = stride_elts;                                       // tensor_dim0_stride lo32
    g1[6] = 0u;
    g1[7] = 0u;
    asm volatile("tensor_load_to_lds %0, %1" :: "s"(g0), "s"(g1) : "memory");
}

__device__ __forceinline__ void wait_tensorcnt0() {
#if __has_builtin(__builtin_amdgcn_s_wait_tensorcnt)
    __builtin_amdgcn_s_wait_tensorcnt(0);
#else
    asm volatile("s_wait_tensorcnt 0x0" ::: "memory");
#endif
}

// ---------------------------------------------------------------------------
// Weight convert+transpose: W f32 [K][N] -> WT f16 [N][K]; 32x32 LDS tiles,
// coalesced reads and writes. Run once per launch, amortized over all GEMMs.
// ---------------------------------------------------------------------------
__global__ __launch_bounds__(256) void wtr_kernel(
    const float* __restrict__ W, f16* __restrict__ WT, int K, int N)
{
    __shared__ f16 Tt[32][40];
    int k0 = blockIdx.x * 32, n0 = blockIdx.y * 32;
    int t = threadIdx.x;
    {
        int r = t >> 3, cq = (t & 7) * 4;     // 32 rows x 8 quads of cols
        float4 v = *(const float4*)(W + (size_t)(k0 + r) * N + n0 + cq);
        Tt[cq + 0][r] = (f16)v.x; Tt[cq + 1][r] = (f16)v.y;
        Tt[cq + 2][r] = (f16)v.z; Tt[cq + 3][r] = (f16)v.w;
    }
    __syncthreads();
    {
        int n = t >> 3, kq = (t & 7) * 4;
        *(v4h*)(WT + (size_t)(n0 + n) * K + k0 + kq) = *(const v4h*)(&Tt[n][kq]);
    }
}

// ---------------------------------------------------------------------------
// LayerNorm (+ optional affine w/b, + optional AdaLN scale/shift from mod+e)
// ---------------------------------------------------------------------------
__global__ __launch_bounds__(256) void modln_kernel(
    const float* __restrict__ x, f16* __restrict__ y,
    const float* __restrict__ mod, const float* __restrict__ evec,
    int shift_idx, int scale_idx,
    const float* __restrict__ w, const float* __restrict__ b,
    int S, int D)
{
    int row = blockIdx.x;
    int bi = row / S;
    const float* xr = x + (size_t)row * D;
    float s = 0.f, ss = 0.f;
    for (int d = threadIdx.x; d < D; d += 256) { float v = xr[d]; s += v; ss += v * v; }
    __shared__ float r0[256], r1[256];
    r0[threadIdx.x] = s; r1[threadIdx.x] = ss;
    __syncthreads();
    for (int off = 128; off > 0; off >>= 1) {
        if (threadIdx.x < off) { r0[threadIdx.x] += r0[threadIdx.x + off]; r1[threadIdx.x] += r1[threadIdx.x + off]; }
        __syncthreads();
    }
    float mean = r0[0] / D;
    float var  = r1[0] / D - mean * mean;
    float inv  = rsqrtf(var + LN_EPS);
    f16* yr = y + (size_t)row * D;
    for (int d = threadIdx.x; d < D; d += 256) {
        float v = (xr[d] - mean) * inv;
        if (w) v = v * w[d] + b[d];
        if (scale_idx >= 0) {
            float sc = 1.f + mod[(size_t)scale_idx * D + d] + evec[((size_t)bi * 6 + scale_idx) * D + d];
            float sh = mod[(size_t)shift_idx * D + d] + evec[((size_t)bi * 6 + shift_idx) * D + d];
            v = v * sc + sh;
        }
        yr[d] = (f16)v;
    }
}

// ---------------------------------------------------------------------------
// RMSNorm over D (+ optional 3-axis RoPE per 128-wide head); f16 output
// ---------------------------------------------------------------------------
__global__ __launch_bounds__(256) void rmsrope_kernel(
    const float* __restrict__ in, f16* __restrict__ out,
    const float* __restrict__ g, const float* __restrict__ theta,
    int S, int D, int do_rope,
    const int* __restrict__ fp, const int* __restrict__ hp, const int* __restrict__ wp)
{
    int row = blockIdx.x;
    const float* xr = in + (size_t)row * D;
    float ss = 0.f;
    for (int d = threadIdx.x; d < D; d += 256) { float v = xr[d]; ss += v * v; }
    __shared__ float red[256];
    red[threadIdx.x] = ss; __syncthreads();
    for (int off = 128; off > 0; off >>= 1) {
        if (threadIdx.x < off) red[threadIdx.x] += red[threadIdx.x + off];
        __syncthreads();
    }
    float inv = rsqrtf(red[0] / D + LN_EPS);
    f16* yr = out + (size_t)row * D;
    if (!do_rope) {
        for (int d = threadIdx.x; d < D; d += 256) yr[d] = (f16)(xr[d] * inv * g[d]);
        return;
    }
    int hd = *hp, wd = *wp;
    (void)fp;
    int sp = row % S;
    int fi = sp / (hd * wd);
    int rem = sp % (hd * wd);
    int hi = rem / wd, wi = rem % wd;
    const int c  = HD / 2;       // 64 pairs per head
    const int c3 = c / 3;        // 21
    const int c0 = c - 2 * c3;   // 22
    int npair = D / 2;
    for (int p = threadIdx.x; p < npair; p += 256) {
        int d0 = 2 * p;
        int ph = (d0 & (HD - 1)) >> 1;
        int idx = (ph < c0) ? fi : (ph < c0 + c3 ? hi : wi);
        float ang = theta[(size_t)idx * c + ph];
        float sn, cs;
        __sincosf(ang, &sn, &cs);
        float x0 = xr[d0] * inv * g[d0];
        float x1 = xr[d0 + 1] * inv * g[d0 + 1];
        yr[d0]     = (f16)(x0 * cs - x1 * sn);
        yr[d0 + 1] = (f16)(x0 * sn + x1 * cs);
    }
}

__global__ void cvt_kernel(const float* __restrict__ in, f16* __restrict__ out, int n) {
    int i = blockIdx.x * 256 + threadIdx.x;
    if (i < n) out[i] = (f16)in[i];
}

// ---------------------------------------------------------------------------
// WMMA GEMM: C[M,N] = A[M,K](f16) * W[K,N] + bias, with epilogues.
// Preferred path: WT (f16, pre-transposed [N][K]) and TDM staging for both
// A and B tiles (padded LDS rows, zero-filled M edge). Fallback (WT==null):
// strided f32->f16 convert staging for B.
// ---------------------------------------------------------------------------
#define BM 128
#define BN 128
#define BK 32
#define KPAD 8

enum { EPI_F32 = 0, EPI_F16 = 1, EPI_GELU_F16 = 2, EPI_RESID_GATE = 3 };

__global__ __launch_bounds__(256) void gemm_kernel(
    const f16* __restrict__ A, const float* __restrict__ W,
    const f16* __restrict__ WT, const float* __restrict__ bias,
    float* __restrict__ outF, f16* __restrict__ outH,
    const float* __restrict__ resid, const float* __restrict__ mod,
    const float* __restrict__ evec, int gate_idx,
    int M, int N, int K, int S, int m_off, int epi)
{
    __shared__ __align__(32) f16 As[BM][BK + KPAD];
    __shared__ __align__(32) f16 Bt[BN][BK + KPAD];   // transposed weight tile [n][k]
    int n0 = blockIdx.x * BN;
    int m0 = blockIdx.y * BM;
    int t  = threadIdx.x;
    int wave = t >> 5, l = t & 31;
    int wr = wave >> 2, wc = wave & 3;
    const unsigned as_lds = (unsigned)(uintptr_t)(void*)&As[0][0];
    const unsigned bt_lds = (unsigned)(uintptr_t)(void*)&Bt[0][0];

    v8f acc[4][2];
#pragma unroll
    for (int r = 0; r < 4; r++)
#pragma unroll
        for (int c = 0; c < 2; c++)
#pragma unroll
            for (int j = 0; j < 8; j++) acc[r][c][j] = 0.f;

    for (int k0 = 0; k0 < K; k0 += BK) {
        // --- TDM: A tile 128 rows x 32 f16 (64B rows -> 80B padded LDS rows);
        //     rows past M read as zero via tensor_dim1 = M - m0.
        if (wave == 0) {
            tdm_load_2d(as_lds, A + (size_t)m0 * K + k0, BK, BM,
                        (unsigned)(M - m0), (unsigned)K, 1, 1, 3, 3);
            if (WT)
                tdm_load_2d(bt_lds, WT + (size_t)n0 * K + k0, BK, BN,
                            BN, (unsigned)K, 1, 1, 3, 3);
        }
        if (!WT) { // fallback: strided convert staging of f32 weights
            int n = t & 127, seg = t >> 7;
            const float* src = W + (size_t)(k0 + seg * 16) * N + n0 + n;
            v16h tmp;
#pragma unroll
            for (int j = 0; j < 16; j++) tmp[j] = (f16)src[(size_t)j * N];
            *(v16h*)(&Bt[n][seg * 16]) = tmp;
        }
        if (wave == 0) wait_tensorcnt0();
        __syncthreads();

        v16h af[4];
#pragma unroll
        for (int r = 0; r < 4; r++) af[r] = a_frag_from(&As[wr * 64 + r * 16][0], BK + KPAD);
#pragma unroll
        for (int c = 0; c < 2; c++) {
            v16h bf = b_frag_nmajor(&Bt[wc * 32 + c * 16][0], BK + KPAD);
#pragma unroll
            for (int r = 0; r < 4; r++)
                acc[r][c] = __builtin_amdgcn_wmma_f32_16x16x32_f16(
                    false, af[r], false, bf, (short)0, acc[r][c], false, false);
        }
        __syncthreads();
    }

    const int grp = l >> 4, ln = l & 15;
#pragma unroll
    for (int r = 0; r < 4; r++)
#pragma unroll
        for (int c = 0; c < 2; c++) {
            int coln = n0 + wc * 32 + c * 16 + ln;
            float bv = bias ? bias[coln] : 0.f;
#pragma unroll
            for (int i = 0; i < 8; i++) {
                int m = m0 + wr * 64 + r * 16 + i + 8 * grp;
                if (m >= M) continue;
                float v = acc[r][c][i] + bv;
                size_t off = (size_t)m * N + coln;
                if (epi == EPI_F32) {
                    outF[off] = v;
                } else if (epi == EPI_F16) {
                    outH[off] = (f16)v;
                } else if (epi == EPI_GELU_F16) {
                    float u = v;
                    float gl = 0.5f * u * (1.f + tanhf(0.7978845608028654f * (u + 0.044715f * u * u * u)));
                    outH[off] = (f16)gl;
                } else { // EPI_RESID_GATE
                    float gmul = 1.f;
                    if (gate_idx >= 0) {
                        int bi = (m + m_off) / S;
                        gmul = mod[(size_t)gate_idx * N + coln] + evec[((size_t)bi * 6 + gate_idx) * N + coln];
                    }
                    outF[off] = resid[off] + v * gmul;
                }
            }
        }
}

// ---------------------------------------------------------------------------
// Flash attention: [B, Seq, NHEADS, HD] f16; online softmax in f32.
// K tile via TDM (TENSORcnt); V tile staged transposed for contiguous B-frags.
// ---------------------------------------------------------------------------
__global__ __launch_bounds__(256) void attn_kernel(
    const f16* __restrict__ Q, const f16* __restrict__ Kv,
    const f16* __restrict__ V, f16* __restrict__ O,
    int Sq, int Sk)
{
    __shared__ __align__(32) f16 Kt[32][HD];        // [key][d], filled by TDM
    __shared__ __align__(32) f16 Vt[HD][32 + KPAD]; // [d][key] transposed
    __shared__ __align__(32) f16 Pt[8][16 * 32];

    int tiles = Sq >> 4;
    int TG = (tiles + 7) >> 3;
    int tg   = blockIdx.x % TG;
    int head = (blockIdx.x / TG) % NHEADS;
    int b    = blockIdx.x / (TG * NHEADS);
    int wave = threadIdx.x >> 5, l = threadIdx.x & 31;
    int qtile = tg * 8 + wave;
    int qt = qtile < tiles ? qtile : tiles - 1;
    const int rstride = NHEADS * HD;
    const unsigned kt_lds = (unsigned)(uintptr_t)(void*)&Kt[0][0];

    const f16* qbase = Q + ((size_t)(b * Sq + qt * 16) * NHEADS + head) * HD;
    v16h aq[4];
#pragma unroll
    for (int c = 0; c < 4; c++) aq[c] = a_frag_from(qbase + c * 32, rstride);

    float m8[8], l8[8];
    v8f o[8];
#pragma unroll
    for (int i = 0; i < 8; i++) { m8[i] = -1e30f; l8[i] = 0.f; }
#pragma unroll
    for (int cc = 0; cc < 8; cc++)
#pragma unroll
        for (int i = 0; i < 8; i++) o[cc][i] = 0.f;

    const float scale = 0.08838834764831845f;  // 1/sqrt(128)
    const int grp = l >> 4, ln = l & 15;

    for (int kb0 = 0; kb0 < Sk; kb0 += 32) {
        if (wave == 0)
            tdm_load_2d(kt_lds, Kv + ((size_t)(b * Sk + kb0) * NHEADS + head) * HD,
                        HD, 32, 32, rstride, 1, 0, 0, 0);
        {
            int d = threadIdx.x & 127, seg = threadIdx.x >> 7;
            const f16* vsrc = V + ((size_t)(b * Sk + kb0 + seg * 16) * NHEADS + head) * HD + d;
            v16h tmp;
#pragma unroll
            for (int j = 0; j < 16; j++) tmp[j] = vsrc[(size_t)j * rstride];
            *(v16h*)(&Vt[d][seg * 16]) = tmp;
        }
        if (wave == 0) wait_tensorcnt0();
        __syncthreads();

        v8f s0, s1;
#pragma unroll
        for (int i = 0; i < 8; i++) { s0[i] = 0.f; s1[i] = 0.f; }
#pragma unroll
        for (int c = 0; c < 4; c++) {
            v16h bk0 = b_frag_nmajor(&Kt[0][c * 32], HD);
            v16h bk1 = b_frag_nmajor(&Kt[16][c * 32], HD);
            s0 = __builtin_amdgcn_wmma_f32_16x16x32_f16(false, aq[c], false, bk0, (short)0, s0, false, false);
            s1 = __builtin_amdgcn_wmma_f32_16x16x32_f16(false, aq[c], false, bk1, (short)0, s1, false, false);
        }

        f16* pw = &Pt[wave][0];
#pragma unroll
        for (int i = 0; i < 8; i++) {
            float v0 = s0[i] * scale, v1 = s1[i] * scale;
            float rm = fmaxf(v0, v1);
            for (int off = 1; off < 16; off <<= 1) rm = fmaxf(rm, __shfl_xor(rm, off, 32));
            float mnew = fmaxf(m8[i], rm);
            float corr = __expf(m8[i] - mnew);
            float p0 = __expf(v0 - mnew);
            float p1 = __expf(v1 - mnew);
            float rsum = p0 + p1;
            for (int off = 1; off < 16; off <<= 1) rsum += __shfl_xor(rsum, off, 32);
            l8[i] = l8[i] * corr + rsum;
            m8[i] = mnew;
#pragma unroll
            for (int cc = 0; cc < 8; cc++) o[cc][i] *= corr;
            int row = i + 8 * grp;
            pw[row * 32 + ln]      = (f16)p0;
            pw[row * 32 + 16 + ln] = (f16)p1;
        }
        asm volatile("s_wait_dscnt 0" ::: "memory");

        v16h pa = a_frag_from(pw, 32);
#pragma unroll
        for (int cc = 0; cc < 8; cc++) {
            v16h bv = b_frag_nmajor(&Vt[cc * 16][0], 32 + KPAD);
            o[cc] = __builtin_amdgcn_wmma_f32_16x16x32_f16(false, pa, false, bv, (short)0, o[cc], false, false);
        }
        __syncthreads();
    }

    if (qtile < tiles) {
        f16* ob = O + ((size_t)(b * Sq + qtile * 16) * NHEADS + head) * HD;
#pragma unroll
        for (int i = 0; i < 8; i++) {
            float inv = 1.f / l8[i];
            int row = i + 8 * grp;
#pragma unroll
            for (int cc = 0; cc < 8; cc++)
                ob[(size_t)row * rstride + cc * 16 + ln] = (f16)(o[cc][i] * inv);
        }
    }
}

// ---------------------------------------------------------------------------
extern "C" void kernel_launch(void* const* d_in, const int* in_sizes, int n_in,
                              void* d_out, int out_size, void* d_ws, size_t ws_size,
                              hipStream_t stream) {
    const float* x      = (const float*)d_in[0];
    const float* e      = (const float*)d_in[1];
    const float* ctx    = (const float*)d_in[2];
    const float* theta  = (const float*)d_in[3];
    const float* mod    = (const float*)d_in[4];
    const float* wq = (const float*)d_in[5];  const float* bq = (const float*)d_in[6];
    const float* wk = (const float*)d_in[7];  const float* bk = (const float*)d_in[8];
    const float* wv = (const float*)d_in[9];  const float* bv = (const float*)d_in[10];
    const float* wo = (const float*)d_in[11]; const float* bo = (const float*)d_in[12];
    const float* nq_w = (const float*)d_in[13]; const float* nk_w = (const float*)d_in[14];
    const float* cwq = (const float*)d_in[15]; const float* cbq = (const float*)d_in[16];
    const float* cwk = (const float*)d_in[17]; const float* cbk = (const float*)d_in[18];
    const float* cwv = (const float*)d_in[19]; const float* cbv = (const float*)d_in[20];
    const float* cwo = (const float*)d_in[21]; const float* cbo = (const float*)d_in[22];
    const float* cnq_w = (const float*)d_in[23]; const float* cnk_w = (const float*)d_in[24];
    const float* n3_w = (const float*)d_in[25]; const float* n3_b = (const float*)d_in[26];
    const float* w1 = (const float*)d_in[27]; const float* b1 = (const float*)d_in[28];
    const float* w2 = (const float*)d_in[29]; const float* b2 = (const float*)d_in[30];
    const int* fptr = (const int*)d_in[31];
    const int* hptr = (const int*)d_in[32];
    const int* wptr = (const int*)d_in[33];

    const int D   = in_sizes[6];               // 1536
    const int B   = in_sizes[1] / (6 * D);     // 2
    const int S   = in_sizes[0] / (B * D);     // 2080
    const int L2  = in_sizes[2] / (B * D);     // 512
    const int Dff = in_sizes[28];              // 8960
    const int M = B * S, Mc = B * L2;
    float* out = (float*)d_out;
    (void)n_in; (void)out_size;

    char* p = (char*)d_ws;
    auto alloc = [&](size_t bytes) -> char* {
        char* r = p; p += (bytes + 255) & ~(size_t)255; return r;
    };
    f16*   y_h   = (f16*)alloc((size_t)M * D * 2);
    f16*   q_h   = (f16*)alloc((size_t)M * D * 2);
    f16*   k_h   = (f16*)alloc((size_t)M * D * 2);
    f16*   v_h   = (f16*)alloc((size_t)M * D * 2);
    f16*   a_h   = (f16*)alloc((size_t)M * D * 2);
    float* bufA  = (float*)alloc((size_t)M * D * 4);
    f16*   ctx_h = (f16*)alloc((size_t)Mc * D * 2);
    f16*   ck_h  = (f16*)alloc((size_t)Mc * D * 2);
    f16*   cv_h  = (f16*)alloc((size_t)Mc * D * 2);
    f16*   hid_h = (f16*)alloc((size_t)S * Dff * 2);   // FFN hidden, one batch at a time

    // Pre-transposed f16 weights if workspace allows (decision is a pure
    // function of ws_size -> deterministic across calls).
    size_t wt_need = ((size_t)8 * D * D + 2 * (size_t)D * Dff) * 2 + 16 * 256;
    bool use_wt = ((size_t)(p - (char*)d_ws) + wt_need) <= ws_size;
    f16 *wq_t = nullptr, *wk_t = nullptr, *wv_t = nullptr, *wo_t = nullptr;
    f16 *cwq_t = nullptr, *cwk_t = nullptr, *cwv_t = nullptr, *cwo_t = nullptr;
    f16 *w1_t = nullptr, *w2_t = nullptr;
    if (use_wt) {
        wq_t  = (f16*)alloc((size_t)D * D * 2);
        wk_t  = (f16*)alloc((size_t)D * D * 2);
        wv_t  = (f16*)alloc((size_t)D * D * 2);
        wo_t  = (f16*)alloc((size_t)D * D * 2);
        cwq_t = (f16*)alloc((size_t)D * D * 2);
        cwk_t = (f16*)alloc((size_t)D * D * 2);
        cwv_t = (f16*)alloc((size_t)D * D * 2);
        cwo_t = (f16*)alloc((size_t)D * D * 2);
        w1_t  = (f16*)alloc((size_t)D * Dff * 2);
        w2_t  = (f16*)alloc((size_t)D * Dff * 2);
        auto wtr = [&](const float* Wm, f16* WTm, int Kk, int Nn) {
            dim3 g(Kk / 32, Nn / 32);
            wtr_kernel<<<g, 256, 0, stream>>>(Wm, WTm, Kk, Nn);
        };
        wtr(wq, wq_t, D, D);   wtr(wk, wk_t, D, D);
        wtr(wv, wv_t, D, D);   wtr(wo, wo_t, D, D);
        wtr(cwq, cwq_t, D, D); wtr(cwk, cwk_t, D, D);
        wtr(cwv, cwv_t, D, D); wtr(cwo, cwo_t, D, D);
        wtr(w1, w1_t, D, Dff); wtr(w2, w2_t, Dff, D);
    }

    auto gemm = [&](const f16* Ah, const float* Wm, const f16* WTm, const float* bs,
                    float* oF, f16* oH, const float* resid, int gate_idx,
                    int Mm, int Nn, int Kk, int m_off, int epi) {
        dim3 g(Nn / BN, (Mm + BM - 1) / BM);
        gemm_kernel<<<g, 256, 0, stream>>>(Ah, Wm, WTm, bs, oF, oH, resid, mod, e,
                                           gate_idx, Mm, Nn, Kk, S, m_off, epi);
    };

    // ---- self-attention ----
    modln_kernel<<<M, 256, 0, stream>>>(x, y_h, mod, e, 0, 1, nullptr, nullptr, S, D);
    gemm(y_h, wq, wq_t, bq, bufA, nullptr, nullptr, -1, M, D, D, 0, EPI_F32);
    rmsrope_kernel<<<M, 256, 0, stream>>>(bufA, q_h, nq_w, theta, S, D, 1, fptr, hptr, wptr);
    gemm(y_h, wk, wk_t, bk, bufA, nullptr, nullptr, -1, M, D, D, 0, EPI_F32);
    rmsrope_kernel<<<M, 256, 0, stream>>>(bufA, k_h, nk_w, theta, S, D, 1, fptr, hptr, wptr);
    gemm(y_h, wv, wv_t, bv, nullptr, v_h, nullptr, -1, M, D, D, 0, EPI_F16);
    {
        int tiles = S / 16, TG = (tiles + 7) / 8;
        attn_kernel<<<B * NHEADS * TG, 256, 0, stream>>>(q_h, k_h, v_h, a_h, S, S);
    }
    gemm(a_h, wo, wo_t, bo, out, nullptr, x, 2, M, D, D, 0, EPI_RESID_GATE);

    // ---- T2V cross-attention ----
    modln_kernel<<<M, 256, 0, stream>>>(out, y_h, mod, e, -1, -1, n3_w, n3_b, S, D);
    gemm(y_h, cwq, cwq_t, cbq, bufA, nullptr, nullptr, -1, M, D, D, 0, EPI_F32);
    rmsrope_kernel<<<M, 256, 0, stream>>>(bufA, q_h, cnq_w, theta, S, D, 0, fptr, hptr, wptr);
    {
        int n = Mc * D;
        cvt_kernel<<<(n + 255) / 256, 256, 0, stream>>>(ctx, ctx_h, n);
    }
    gemm(ctx_h, cwk, cwk_t, cbk, bufA, nullptr, nullptr, -1, Mc, D, D, 0, EPI_F32);
    rmsrope_kernel<<<Mc, 256, 0, stream>>>(bufA, ck_h, cnk_w, theta, L2, D, 0, fptr, hptr, wptr);
    gemm(ctx_h, cwv, cwv_t, cbv, nullptr, cv_h, nullptr, -1, Mc, D, D, 0, EPI_F16);
    {
        int tiles = S / 16, TG = (tiles + 7) / 8;
        attn_kernel<<<B * NHEADS * TG, 256, 0, stream>>>(q_h, ck_h, cv_h, a_h, S, L2);
    }
    gemm(a_h, cwo, cwo_t, cbo, out, nullptr, out, -1, M, D, D, 0, EPI_RESID_GATE);

    // ---- modulated FFN (chunked per batch to bound hidden scratch) ----
    modln_kernel<<<M, 256, 0, stream>>>(out, y_h, mod, e, 3, 4, nullptr, nullptr, S, D);
    for (int bb = 0; bb < B; bb++) {
        const f16* yb = y_h + (size_t)bb * S * D;
        float* ob = out + (size_t)bb * S * D;
        gemm(yb, w1, w1_t, b1, nullptr, hid_h, nullptr, -1, S, Dff, D, 0, EPI_GELU_F16);
        gemm(hid_h, w2, w2_t, b2, ob, nullptr, ob, 5, S, D, Dff, bb * S, EPI_RESID_GATE);
    }
}
